// MultiHeadAttention_34308198761050
// MI455X (gfx1250) — matrix operand order repaired and checked by reading, not compiled, not run on hardware
//
#include <hip/hip_runtime.h>

#define L_SEQ 4096
#define DMODEL 768
#define NHEADS 12
#define DK 64

typedef __attribute__((ext_vector_type(16))) _Float16 v16h;
typedef __attribute__((ext_vector_type(8)))  float    v8f;
typedef __attribute__((ext_vector_type(4)))  int      v4i;

union H16 { v16h v; _Float16 h[16]; uint4 u[2]; };

// ---------- CDNA5 async global->LDS copy (ASYNCcnt path) ----------
#if defined(__gfx1250__) && __has_builtin(__builtin_amdgcn_global_load_async_to_lds_b128)
#define HAVE_ASYNC_LDS 1
typedef __attribute__((address_space(1))) v4i* g4p;  // global v4i*
typedef __attribute__((address_space(3))) v4i* l4p;  // LDS v4i*
#else
#define HAVE_ASYNC_LDS 0
#endif

__device__ __forceinline__ void copy16_to_lds(_Float16* l, const _Float16* g) {
#if HAVE_ASYNC_LDS
  __builtin_amdgcn_global_load_async_to_lds_b128((g4p)(void*)g, (l4p)(void*)l, 0, 0);
#else
  *(uint4*)l = *(const uint4*)g;
#endif
}

__device__ __forceinline__ void async_wait() {
#if HAVE_ASYNC_LDS
#if __has_builtin(__builtin_amdgcn_s_wait_asynccnt)
  __builtin_amdgcn_s_wait_asynccnt(0);
#else
  asm volatile("s_wait_asynccnt 0x0" ::: "memory");
#endif
#endif
}

// ---------- WMMA helpers ----------
__device__ __forceinline__ v8f wmma32(v16h a, v16h b, v8f c) {
  // D = A(16x32 f16) * B(32x16 f16) + C(16x16 f32)
  return __builtin_amdgcn_wmma_f32_16x16x32_f16(false, a, false, b, (short)0, c,
                                                false, false);
}

// A-matrix 16x32 f16 fragment (lane m=l&15; VGPR0-3 k=8*half+0..7,
// VGPR4-7 k=16+8*half+0..7) -> two b128 loads per lane.
__device__ __forceinline__ v16h frag_a(const _Float16* base, int row0, int stride,
                                       int kbase) {
  const int lane = threadIdx.x & 31;
  const int m = lane & 15, hh = lane >> 4;
  const _Float16* p = base + (size_t)(row0 + m) * stride + kbase + 8 * hh;
  H16 f;
  f.u[0] = *(const uint4*)p;
  f.u[1] = *(const uint4*)(p + 16);
  return f.v;
}

// B-matrix 32x16 f16 fragment (lane n=l&15; VGPR0-3 k=8*half+0..7, VGPR4-7 k=16+8*half+0..7)
// -> two b128 loads per lane.
__device__ __forceinline__ v16h frag_b(const _Float16* base, int col0, int stride,
                                       int kbase) {
  const int lane = threadIdx.x & 31;
  const int nn = lane & 15, hh = lane >> 4;
  const _Float16* p = base + (size_t)(col0 + nn) * stride + kbase + 8 * hh;
  H16 f;
  f.u[0] = *(const uint4*)p;
  f.u[1] = *(const uint4*)(p + 16);
  return f.v;
}

// ---------- projection GEMM: C = X @ W^T + bias, M=4096 N=768 K=768 ----------
// MODE 0: X f32, out f16 as Qh/Kh [H][L][64]
// MODE 1: X f32, out f16 as Vt [H][64][L]  (== flat [col][row])
// MODE 2: X f16 (attn), out f32 row-major [L][768]
template <int MODE>
__global__ __launch_bounds__(256) void gemm_proj_kernel(
    const float* __restrict__ Xf, const _Float16* __restrict__ Xh,
    const float* __restrict__ W, const float* __restrict__ bias,
    _Float16* __restrict__ outh, float* __restrict__ outf) {
  __shared__ __align__(16) _Float16 A_lds[2][128 * 32];
  __shared__ __align__(16) _Float16 B_lds[2][64 * 32];

  const int m0 = blockIdx.x * 128;
  const int n0 = blockIdx.y * 64;
  const int t = threadIdx.x;
  const int wv = t >> 5;
  const int lane = t & 31;
  const int n = lane & 15;
  const int hh = lane >> 4;

  const int arow = t >> 1, akk = (t & 1) * 16;   // A stage: 128 rows x 32 k
  const int brow = t >> 2, bkk = (t & 3) * 8;    // B stage: 64 rows x 32 k

  v8f acc[4];
#pragma unroll
  for (int nb = 0; nb < 4; ++nb)
#pragma unroll
    for (int r = 0; r < 8; ++r) acc[nb][r] = 0.0f;

  // ---- stage one K-chunk (f32 -> f16 convert in flight) ----
  auto stage = [&](int k0, int buf) {
    H16 hv;
    if (MODE == 2) {
      const _Float16* src = Xh + (size_t)(m0 + arow) * DMODEL + k0 + akk;
      hv.u[0] = ((const uint4*)src)[0];
      hv.u[1] = ((const uint4*)src)[1];
    } else {
      const float4* s4 = (const float4*)(Xf + (size_t)(m0 + arow) * DMODEL + k0 + akk);
      float4 f0 = s4[0], f1 = s4[1], f2 = s4[2], f3 = s4[3];
      float tf[16] = {f0.x, f0.y, f0.z, f0.w, f1.x, f1.y, f1.z, f1.w,
                      f2.x, f2.y, f2.z, f2.w, f3.x, f3.y, f3.z, f3.w};
#pragma unroll
      for (int i = 0; i < 16; ++i) hv.h[i] = (_Float16)tf[i];
    }
    *(uint4*)&A_lds[buf][arow * 32 + akk] = hv.u[0];
    *(uint4*)&A_lds[buf][arow * 32 + akk + 8] = hv.u[1];

    const float4* s4 = (const float4*)(W + (size_t)(n0 + brow) * DMODEL + k0 + bkk);
    float4 f0 = s4[0], f1 = s4[1];
    float tf[8] = {f0.x, f0.y, f0.z, f0.w, f1.x, f1.y, f1.z, f1.w};
    H16 hb;
#pragma unroll
    for (int i = 0; i < 8; ++i) hb.h[i] = (_Float16)tf[i];
    *(uint4*)&B_lds[buf][brow * 32 + bkk] = hb.u[0];
  };

  stage(0, 0);
  __syncthreads();

  int cur = 0;
  for (int k0 = 0; k0 < DMODEL; k0 += 32) {
    if (k0 + 32 < DMODEL) stage(k0 + 32, cur ^ 1);  // overlap with WMMAs below

    // all fragments first, then the WMMA group (batched dscnt waits)
    const v16h a = frag_a(A_lds[cur], wv * 16, 32, 0);
    const v16h b0 = frag_b(B_lds[cur], 0, 32, 0);
    const v16h b1 = frag_b(B_lds[cur], 16, 32, 0);
    const v16h b2 = frag_b(B_lds[cur], 32, 32, 0);
    const v16h b3 = frag_b(B_lds[cur], 48, 32, 0);
    acc[0] = wmma32(a, b0, acc[0]);
    acc[1] = wmma32(a, b1, acc[1]);
    acc[2] = wmma32(a, b2, acc[2]);
    acc[3] = wmma32(a, b3, acc[3]);

    __syncthreads();
    cur ^= 1;
  }

  // ---- epilogue: bias + layout-specific store ----
#pragma unroll
  for (int nb = 0; nb < 4; ++nb) {
    const int col = n0 + nb * 16 + n;
    const float bb = bias[col];
#pragma unroll
    for (int r = 0; r < 8; ++r) {
      const int gr = m0 + wv * 16 + r + 8 * hh;
      const float val = acc[nb][r] + bb;
      if (MODE == 0) {
        const int h = col >> 6, d = col & 63;
        outh[((size_t)h * L_SEQ + gr) * DK + d] = (_Float16)val;
      } else if (MODE == 1) {
        outh[(size_t)col * L_SEQ + gr] = (_Float16)val;
      } else {
        outf[(size_t)gr * DMODEL + col] = val;
      }
    }
  }
}

// ---------- flash attention ----------
// block = (128 query rows, head); wave owns 16 rows; keys streamed in chunks
// of 32 with online softmax. Qh/Kh: [H][L][64] f16, Vt: [H][64][L] f16,
// attn out: [L][768] f16. K/V chunks double-buffered via async global->LDS.
__global__ __launch_bounds__(256) void flash_attn_kernel(
    const _Float16* __restrict__ Qh, const _Float16* __restrict__ Kh,
    const _Float16* __restrict__ Vt, _Float16* __restrict__ attn) {
  __shared__ __align__(16) _Float16 K_lds[2][32 * 64];
  __shared__ __align__(16) _Float16 V_lds[2][64 * 32];
  __shared__ __align__(16) _Float16 P_lds[8 * 512];

  const int head = blockIdx.y;
  const int q0 = blockIdx.x * 128;
  const int t = threadIdx.x;
  const int wv = t >> 5;
  const int lane = t & 31;
  const int n = lane & 15;
  const int hh = lane >> 4;
  const float scale = 0.125f;  // 1/sqrt(64)

  const _Float16* Qg = Qh + ((size_t)head * L_SEQ + q0 + wv * 16) * DK;
  const _Float16* Kg = Kh + (size_t)head * L_SEQ * DK;
  const _Float16* Vg = Vt + (size_t)head * DK * L_SEQ;

  const int kr = t >> 3, kc = (t & 7) * 8;  // K stage: 32 x 64
  const int vr = t >> 2, vc = (t & 3) * 8;  // V stage: 64 x 32

  auto stage = [&](int j0, int buf) {
    copy16_to_lds(&K_lds[buf][kr * 64 + kc], Kg + (size_t)(j0 + kr) * DK + kc);
    copy16_to_lds(&V_lds[buf][vr * 32 + vc], Vg + (size_t)vr * L_SEQ + j0 + vc);
  };

  // Q fragments for this wave's 16 rows, loaded once.
  const v16h aq0 = frag_a(Qg, 0, DK, 0);
  const v16h aq1 = frag_a(Qg, 0, DK, 32);

  float mi[8], li[8];
  v8f o[4];
#pragma unroll
  for (int r = 0; r < 8; ++r) { mi[r] = -1e30f; li[r] = 0.0f; }
#pragma unroll
  for (int nb = 0; nb < 4; ++nb)
#pragma unroll
    for (int r = 0; r < 8; ++r) o[nb][r] = 0.0f;

  stage(0, 0);
  async_wait();
  __syncthreads();

  int cur = 0;
  for (int j0 = 0; j0 < L_SEQ; j0 += 32) {
    if (j0 + 32 < L_SEQ) stage(j0 + 32, cur ^ 1);  // async, overlaps compute

    // ---- scores S(16x32) = Q(16x64) @ K_chunk^T : 4 WMMAs ----
    const v16h bl0 = frag_b(K_lds[cur], 0, 64, 0);
    const v16h bh0 = frag_b(K_lds[cur], 0, 64, 32);
    const v16h bl1 = frag_b(K_lds[cur], 16, 64, 0);
    const v16h bh1 = frag_b(K_lds[cur], 16, 64, 32);
    v8f s0, s1;
#pragma unroll
    for (int r = 0; r < 8; ++r) { s0[r] = 0.0f; s1[r] = 0.0f; }
    s0 = wmma32(aq0, bl0, s0);
    s0 = wmma32(aq1, bh0, s0);
    s1 = wmma32(aq0, bl1, s1);
    s1 = wmma32(aq1, bh1, s1);
#pragma unroll
    for (int r = 0; r < 8; ++r) { s0[r] *= scale; s1[r] *= scale; }

    // ---- online softmax over the 32-key chunk ----
    float mnew[8], alpha[8];
#pragma unroll
    for (int r = 0; r < 8; ++r) {
      float mx = fmaxf(s0[r], s1[r]);
      mx = fmaxf(mx, __shfl_xor(mx, 1, 32));
      mx = fmaxf(mx, __shfl_xor(mx, 2, 32));
      mx = fmaxf(mx, __shfl_xor(mx, 4, 32));
      mx = fmaxf(mx, __shfl_xor(mx, 8, 32));
      mnew[r] = fmaxf(mi[r], mx);
      alpha[r] = __expf(mi[r] - mnew[r]);
      mi[r] = mnew[r];
    }
#pragma unroll
    for (int r = 0; r < 8; ++r) {
      s0[r] = __expf(s0[r] - mnew[r]);
      s1[r] = __expf(s1[r] - mnew[r]);
    }
#pragma unroll
    for (int r = 0; r < 8; ++r) {
      float rs = s0[r] + s1[r];
      rs += __shfl_xor(rs, 1, 32);
      rs += __shfl_xor(rs, 2, 32);
      rs += __shfl_xor(rs, 4, 32);
      rs += __shfl_xor(rs, 8, 32);
      li[r] = li[r] * alpha[r] + rs;
    }
#pragma unroll
    for (int nb = 0; nb < 4; ++nb)
#pragma unroll
      for (int r = 0; r < 8; ++r) o[nb][r] *= alpha[r];

    // ---- P (C-layout f32) -> per-wave LDS tile -> A-layout f16 fragment ----
    _Float16* Pw = &P_lds[wv * 512];
#pragma unroll
    for (int r = 0; r < 8; ++r) {
      Pw[(r + 8 * hh) * 32 + n] = (_Float16)s0[r];
      Pw[(r + 8 * hh) * 32 + 16 + n] = (_Float16)s1[r];
    }
    const v16h ap = frag_a(Pw, 0, 32, 0);

    // ---- O(16x64) += P(16x32) @ V_chunk(32x64) : 4 WMMAs ----
    const v16h bv0 = frag_b(V_lds[cur], 0, 32, 0);
    const v16h bv1 = frag_b(V_lds[cur], 16, 32, 0);
    const v16h bv2 = frag_b(V_lds[cur], 32, 32, 0);
    const v16h bv3 = frag_b(V_lds[cur], 48, 32, 0);
    o[0] = wmma32(ap, bv0, o[0]);
    o[1] = wmma32(ap, bv1, o[1]);
    o[2] = wmma32(ap, bv2, o[2]);
    o[3] = wmma32(ap, bv3, o[3]);

    async_wait();      // next chunk's async LDS writes landed
    __syncthreads();   // everyone done with cur before it becomes the write buf
    cur ^= 1;
  }

  // ---- normalize and store f16 attn output [L][768] ----
  const int qrow = q0 + wv * 16;
#pragma unroll
  for (int r = 0; r < 8; ++r) {
    const float inv = 1.0f / li[r];
#pragma unroll
    for (int nb = 0; nb < 4; ++nb) {
      attn[(size_t)(qrow + r + 8 * hh) * DMODEL + head * DK + nb * 16 + n] =
          (_Float16)(o[nb][r] * inv);
    }
  }
}

extern "C" void kernel_launch(void* const* d_in, const int* in_sizes, int n_in,
                              void* d_out, int out_size, void* d_ws, size_t ws_size,
                              hipStream_t stream) {
  const float* q  = (const float*)d_in[0];
  const float* k  = (const float*)d_in[1];
  const float* v  = (const float*)d_in[2];
  const float* Wq = (const float*)d_in[3];
  const float* bq = (const float*)d_in[4];
  const float* Wk = (const float*)d_in[5];
  const float* bk = (const float*)d_in[6];
  const float* Wv = (const float*)d_in[7];
  const float* bv = (const float*)d_in[8];
  const float* Wo = (const float*)d_in[9];
  const float* bo = (const float*)d_in[10];

  _Float16* ws = (_Float16*)d_ws;
  const size_t HLD = (size_t)NHEADS * L_SEQ * DK;  // 3,145,728 elems
  _Float16* Qh = ws;
  _Float16* Kh = ws + HLD;
  _Float16* Vt = ws + 2 * HLD;
  _Float16* At = ws + 3 * HLD;  // total ws use: 24 MB f16

  dim3 grid(L_SEQ / 128, DMODEL / 64);  // 32 x 12
  dim3 blk(256);

  gemm_proj_kernel<0><<<grid, blk, 0, stream>>>(q, nullptr, Wq, bq, Qh, nullptr);
  gemm_proj_kernel<0><<<grid, blk, 0, stream>>>(k, nullptr, Wk, bk, Kh, nullptr);
  gemm_proj_kernel<1><<<grid, blk, 0, stream>>>(v, nullptr, Wv, bv, Vt, nullptr);
  flash_attn_kernel<<<dim3(L_SEQ / 128, NHEADS), blk, 0, stream>>>(Qh, Kh, Vt, At);
  gemm_proj_kernel<2><<<grid, blk, 0, stream>>>(nullptr, At, Wo, bo, nullptr,
                                                (float*)d_out);
}
